// PlanarNormalizingFlow_57681410786049
// MI455X (gfx1250) — matrix operand order repaired and checked by reading, not compiled, not run on hardware
//
#include <hip/hip_runtime.h>
#include <hip/hip_bf16.h>

// ---------------------------------------------------------------------------
// PlanarNormalizingFlow + 3-means IM penalty, tuned for MI455X (gfx1250).
//
//   * one-time f32->bf16 conversion of z (128 MB < 192 MB L2) so the 30
//     k-means assignment passes run out of L2 at bf16 width
//   * z @ [c0 c1 c2 w]^T tiles via v_wmma_f32_16x16x32_bf16 (wave32 WMMA)
//   * assign kernel phased so cluster accumulators are never live across the
//     WMMA pipeline (kills the scratch spills seen in round 1)
//   * parallel 771-block partial reduction so no single workgroup ever has
//     to stream MBs serially; deterministic fixed-order sums throughout
//   * final pass streams f32 z exactly once for f_z / logdet / penalty
// ---------------------------------------------------------------------------

#define BROWS 262144
#define DDIM  256
#define NBLK_ASSIGN 1024   // 8 waves/block * 2 tiles/wave * 16 rows = 16384 tiles
#define NBLK_FINAL  2048   // 8 waves/block * 1 tile/wave
#define PARTW 772          // 3*256 sums + 3 counts + pad
#define NRED  771          // reduced elements (768 sums + 3 counts)

typedef __attribute__((ext_vector_type(16))) __bf16    v16bf;
typedef __attribute__((ext_vector_type(8)))  float     v8f;
typedef __attribute__((ext_vector_type(4)))  unsigned int v4u;

union FragBF { v4u u[2]; v16bf f; };
union P8     { v4u u; __hip_bfloat16 h[8]; };

// ---- WMMA fragment loaders (per ISA 16-bit A/B VGPR layouts, wave32) ------
__device__ __forceinline__ v16bf load_frag_A(const __hip_bfloat16* zb, int r0,
                                             int kb, int lane) {
  int half = lane >> 4, m = lane & 15;
  const __hip_bfloat16* p = zb + (size_t)(r0 + m) * DDIM + kb + half * 8;
  FragBF f;
  f.u[0] = *(const v4u*)p;         // K = kb+8h .. +7
  f.u[1] = *(const v4u*)(p + 16);  // K = kb+16+8h .. +7
  return f.f;
}

__device__ __forceinline__ v16bf load_frag_B(const __hip_bfloat16* cmat,
                                             int kb, int lane) {
  int half = lane >> 4, n = lane & 15;
  const __hip_bfloat16* p = cmat + n * DDIM + kb + half * 8;
  FragBF f;
  f.u[0] = *(const v4u*)p;         // K = kb+8h .. +7
  f.u[1] = *(const v4u*)(p + 16);  // K = kb+16+8h .. +7
  return f.f;
}

// acc[v] = dot(z[r0 + v + 8*(lane>>4)], col(lane&15)) over K=256
__device__ __forceinline__ v8f dots_tile(const __hip_bfloat16* zb,
                                         const __hip_bfloat16* cmat,
                                         int r0, int lane) {
  v8f acc = {};
#pragma unroll
  for (int kb = 0; kb < DDIM; kb += 32) {
    v16bf a  = load_frag_A(zb, r0, kb, lane);
    v16bf bm = load_frag_B(cmat, kb, lane);
    acc = __builtin_amdgcn_wmma_f32_16x16x32_bf16(false, a, false, bm,
                                                  (short)0, acc, false, false);
  }
  return acc;
}

// ---- block-wide sum over 256 threads (deterministic, result broadcast) ----
__device__ __forceinline__ float block_reduce_sum(float v, float* sred, int t) {
  sred[t] = v; __syncthreads();
  for (int s = 128; s > 0; s >>= 1) {
    if (t < s) sred[t] += sred[t + s];
    __syncthreads();
  }
  float r = sred[0]; __syncthreads();
  return r;
}

// ---------------------------------------------------------------------------
__global__ void convert_kernel(const float* __restrict__ z,
                               __hip_bfloat16* __restrict__ zb) {
  size_t i = ((size_t)blockIdx.x * 256 + threadIdx.x) * 8;
  const float4* zp = (const float4*)(z + i);
  float4 a = zp[0], c = zp[1];
  P8 p;
  p.h[0] = __float2bfloat16(a.x); p.h[1] = __float2bfloat16(a.y);
  p.h[2] = __float2bfloat16(a.z); p.h[3] = __float2bfloat16(a.w);
  p.h[4] = __float2bfloat16(c.x); p.h[5] = __float2bfloat16(c.y);
  p.h[6] = __float2bfloat16(c.z); p.h[7] = __float2bfloat16(c.w);
  *(v4u*)(zb + i) = p.u;
}

// deterministic stand-in for jax top_k(uniform(key 42)) row choice
__global__ void init_kernel(const float* __restrict__ z, float* centers,
                            __hip_bfloat16* cmat, float* cn2, float* shift) {
  __shared__ float sred[256];
  int t = threadIdx.x;
  const int IND[3] = {97, 131072, 262143};
#pragma unroll
  for (int c = 0; c < 3; ++c) {
    float v = z[(size_t)IND[c] * DDIM + t];
    centers[c * DDIM + t] = v;
    cmat[c * DDIM + t] = __float2bfloat16(v);
    float n2 = block_reduce_sum(v * v, sred, t);
    if (t == 0) cn2[c] = n2;
  }
#pragma unroll
  for (int r = 3; r < 16; ++r) cmat[r * DDIM + t] = __float2bfloat16(0.f);
  if (t == 0) shift[0] = 100.f;
}

// assignment + per-block sufficient statistics (one k-means half-step).
// Phase A: WMMA dots + argmin (no heavy live state).
// Phase B: per-lane accumulation (no WMMA live) -> no spills.
__global__ void assign_kernel(const __hip_bfloat16* __restrict__ zb,
                              const __hip_bfloat16* __restrict__ cmat,
                              const float* __restrict__ cn2,
                              float* __restrict__ partial) {
  __shared__ float sdots[8][16][4];
  __shared__ int   sbest[8][2][16];
  __shared__ int   scnt[3];
  __shared__ float swacc[8][768];
  int t = threadIdx.x, w = t >> 5, lane = t & 31;
  int half = lane >> 4, col = lane & 15;

  if (t < 3) scnt[t] = 0;
  __syncthreads();

  float c0 = cn2[0], c1 = cn2[1], c2 = cn2[2];

  // ---- Phase A: dots via WMMA, per-row argmin -----------------------------
  for (int tt = 0; tt < 2; ++tt) {
    int r0 = ((blockIdx.x * 8 + w) * 2 + tt) * 16;
    v8f acc = dots_tile(zb, cmat, r0, lane);     // EXEC all-1s here
    if (col < 3) {
#pragma unroll
      for (int v = 0; v < 8; ++v) sdots[w][v + 8 * half][col] = acc[v];
    }
    __syncthreads();
    if (lane < 16) {
      float d0 = c0 - 2.f * sdots[w][lane][0];
      float d1 = c1 - 2.f * sdots[w][lane][1];
      float d2 = c2 - 2.f * sdots[w][lane][2];
      int bj = 0; float bd = d0;
      if (d1 < bd) { bd = d1; bj = 1; }
      if (d2 < bd) { bd = d2; bj = 2; }
      sbest[w][tt][lane] = bj;
      atomicAdd(&scnt[bj], 1);                   // int: order-exact
    }
    __syncthreads();
  }

  // ---- Phase B: accumulate rows; lane owns dims [lane*8, lane*8+8) --------
  float f0[8], f1[8], f2[8];
#pragma unroll
  for (int i = 0; i < 8; ++i) { f0[i] = 0.f; f1[i] = 0.f; f2[i] = 0.f; }

  for (int tt = 0; tt < 2; ++tt) {
    int r0 = ((blockIdx.x * 8 + w) * 2 + tt) * 16;
#pragma unroll 4
    for (int r = 0; r < 16; ++r) {
      int bj = sbest[w][tt][r];                  // wave-uniform
      P8 p; p.u = *(const v4u*)(zb + (size_t)(r0 + r) * DDIM + lane * 8);
      float zv[8];
#pragma unroll
      for (int i = 0; i < 8; ++i) zv[i] = __bfloat162float(p.h[i]);
      if (bj == 0) {
#pragma unroll
        for (int i = 0; i < 8; ++i) f0[i] += zv[i];
      } else if (bj == 1) {
#pragma unroll
        for (int i = 0; i < 8; ++i) f1[i] += zv[i];
      } else {
#pragma unroll
        for (int i = 0; i < 8; ++i) f2[i] += zv[i];
      }
    }
  }

  // ---- fixed-order block combine (deterministic, no float atomics) -------
#pragma unroll
  for (int i = 0; i < 8; ++i) {
    swacc[w][        lane * 8 + i] = f0[i];
    swacc[w][256 +   lane * 8 + i] = f1[i];
    swacc[w][512 +   lane * 8 + i] = f2[i];
  }
  __syncthreads();
  float s0 = 0.f, s1 = 0.f, s2 = 0.f;
#pragma unroll
  for (int wv = 0; wv < 8; ++wv) {
    s0 += swacc[wv][t]; s1 += swacc[wv][256 + t]; s2 += swacc[wv][512 + t];
  }
  float* pp = partial + (size_t)blockIdx.x * PARTW;
  pp[t] = s0; pp[256 + t] = s1; pp[512 + t] = s2;
  if (t < 3) pp[768 + t] = (float)scnt[t];
}

// parallel reduction of per-block partials: one block per output element
__global__ void reduce_kernel(const float* __restrict__ partial,
                              float* __restrict__ reduced) {
  __shared__ float sred[256];
  int e = blockIdx.x, t = threadIdx.x;
  float s = 0.f;
  for (int b = t; b < NBLK_ASSIGN; b += 256)
    s += partial[(size_t)b * PARTW + e];
  float r = block_reduce_sum(s, sred, t);
  if (t == 0) reduced[e] = r;
}

// apply means with convergence-freeze, refresh bf16 B-matrix + norms
__global__ void update_kernel(const float* __restrict__ reduced, float* centers,
                              __hip_bfloat16* cmat, float* cn2, float* shift) {
  __shared__ float sred[256];
  __shared__ float sss[3];
  __shared__ int   sact;
  int t = threadIdx.x;
  float mean[3], oldc[3];
#pragma unroll
  for (int c = 0; c < 3; ++c) {
    float cnt = reduced[768 + c];
    mean[c] = reduced[c * 256 + t] / fmaxf(cnt, 1.0f);
    oldc[c] = centers[c * DDIM + t];
    float diff = mean[c] - oldc[c];
    float ss = block_reduce_sum(diff * diff, sred, t);
    if (t == 0) sss[c] = ss;
  }
  __syncthreads();
  if (t == 0) {
    float sh = shift[0];
    int active = (sh * sh >= 1e-4f) ? 1 : 0;
    if (active) {
      float ns = 0.f;
#pragma unroll
      for (int c = 0; c < 3; ++c) if (sss[c] > 0.f) ns += sqrtf(sss[c]);
      shift[0] = ns;
    }
    sact = active;
  }
  __syncthreads();
#pragma unroll
  for (int c = 0; c < 3; ++c) {
    float cv = sact ? mean[c] : oldc[c];
    centers[c * DDIM + t] = cv;
    cmat[c * DDIM + t] = __float2bfloat16(cv);
    float n2 = block_reduce_sum(cv * cv, sred, t);
    if (t == 0) cn2[c] = n2;
  }
}

// uhat / muw prep; packs w as column 3 of the WMMA B-matrix
__global__ void prep_kernel(const float* __restrict__ u, const float* __restrict__ w,
                            __hip_bfloat16* cmat, float* uhat, float* wuhat) {
  __shared__ float sred[256];
  __shared__ float scoef;
  int t = threadIdx.x;
  float uv = u[t], wv = w[t];
  float uw = block_reduce_sum(uv * wv, sred, t);
  float ww = block_reduce_sum(wv * wv, sred, t);
  if (t == 0) {
    float sp  = fmaxf(uw, 0.f) + log1pf(expf(-fabsf(uw)));  // softplus
    float muw = -1.f + sp;
    scoef = (muw - uw) / ww;
  }
  __syncthreads();
  float uh = uv + scoef * wv;
  uhat[t] = uh;
  cmat[3 * DDIM + t] = __float2bfloat16(wv);
  float wu = block_reduce_sum(wv * uh, sred, t);
  if (t == 0) wuhat[0] = wu;
}

// single streaming pass: f_z, logdet, penalty
__global__ void final_kernel(const float* __restrict__ z,
                             const __hip_bfloat16* __restrict__ zb,
                             const __hip_bfloat16* __restrict__ cmat,
                             const float* __restrict__ cn2,
                             const float* __restrict__ uhat,
                             const float* __restrict__ wuhat,
                             const float* __restrict__ bptr,
                             float* __restrict__ out) {
  __shared__ float suh[256];
  __shared__ float sdots[8][16][4];
  int t = threadIdx.x, w = t >> 5, lane = t & 31;
  int half = lane >> 4, col = lane & 15;
  suh[t] = uhat[t];
  __syncthreads();

  int r0 = (blockIdx.x * 8 + w) * 16;
  v8f acc = dots_tile(zb, cmat, r0, lane);       // cols: c0,c1,c2,w
  if (col < 4) {
#pragma unroll
    for (int v = 0; v < 8; ++v) sdots[w][v + 8 * half][col] = acc[v];
  }
  __syncthreads();

  float bval = bptr[0];
  float wu   = wuhat[0];
  float c0 = cn2[0], c1 = cn2[1], c2 = cn2[2];
  float* logdet = out + (size_t)BROWS * DDIM;
  float* pen    = logdet + BROWS;

  for (int r = 0; r < 16; ++r) {
    int row = r0 + r;
    float d3 = sdots[w][r][3];
    float th = tanhf(d3 + bval);
    const float* zr = z + (size_t)row * DDIM + lane * 8;
    float4 a  = *(const float4*)zr;
    float4 b4 = *(const float4*)(zr + 4);
    float* fr = out + (size_t)row * DDIM + lane * 8;
    float4 o1, o2;
    o1.x = a.x  + suh[lane * 8 + 0] * th; o1.y = a.y  + suh[lane * 8 + 1] * th;
    o1.z = a.z  + suh[lane * 8 + 2] * th; o1.w = a.w  + suh[lane * 8 + 3] * th;
    o2.x = b4.x + suh[lane * 8 + 4] * th; o2.y = b4.y + suh[lane * 8 + 5] * th;
    o2.z = b4.z + suh[lane * 8 + 6] * th; o2.w = b4.w + suh[lane * 8 + 7] * th;
    *(float4*)fr = o1; *(float4*)(fr + 4) = o2;

    float sq = a.x*a.x + a.y*a.y + a.z*a.z + a.w*a.w
             + b4.x*b4.x + b4.y*b4.y + b4.z*b4.z + b4.w*b4.w;
#pragma unroll
    for (int m = 16; m > 0; m >>= 1) sq += __shfl_xor(sq, m);
    if (lane == 0) {
      float dmin = fminf(fminf(c0 - 2.f * sdots[w][r][0],
                               c1 - 2.f * sdots[w][r][1]),
                               c2 - 2.f * sdots[w][r][2]);
      float minsq = fmaxf(sq + dmin, 0.f);
      const float S[7] = {0.1f, 0.2f, 0.5f, 1.f, 2.f, 5.f, 10.f};
      float beta = 0.f;
#pragma unroll
      for (int s = 0; s < 7; ++s) { float C = 2.f * DDIM * S[s]; beta += C / (C + minsq); }
      float psi = (1.f - th * th) * wu;
      logdet[row] = logf(fabsf(1.f + psi) + 1e-7f);
      pen[row]    = 0.01f * logf(fabsf(beta) + 1e-7f);
    }
  }
}

// ---------------------------------------------------------------------------
extern "C" void kernel_launch(void* const* d_in, const int* in_sizes, int n_in,
                              void* d_out, int out_size, void* d_ws, size_t ws_size,
                              hipStream_t stream) {
  (void)in_sizes; (void)n_in; (void)out_size; (void)ws_size;
  const float* z = (const float*)d_in[0];
  const float* u = (const float*)d_in[1];
  const float* w = (const float*)d_in[2];
  const float* b = (const float*)d_in[3];

  char* ws = (char*)d_ws;
  const size_t ZB_BYTES = (size_t)BROWS * DDIM * 2;          // 128 MB bf16 copy
  __hip_bfloat16* zb   = (__hip_bfloat16*)ws;
  float* centers       = (float*)(ws + ZB_BYTES);            // 3*256 f32
  float* cn2           = (float*)(ws + ZB_BYTES + 3072);     // 16 f32
  float* shift         = (float*)(ws + ZB_BYTES + 3072 + 64);
  __hip_bfloat16* cmat = (__hip_bfloat16*)(ws + ZB_BYTES + 4096);    // 16x256 bf16
  float* uhat          = (float*)(ws + ZB_BYTES + 12288);
  float* wuhat         = (float*)(ws + ZB_BYTES + 13312);
  float* reduced       = (float*)(ws + ZB_BYTES + 16384);    // 771 f32
  float* partial       = (float*)(ws + ZB_BYTES + 20480);    // 1024*772 f32

  convert_kernel<<<(BROWS * (DDIM / 8)) / 256, 256, 0, stream>>>(z, zb);
  init_kernel<<<1, 256, 0, stream>>>(z, centers, cmat, cn2, shift);
  for (int it = 0; it < 30; ++it) {
    assign_kernel<<<NBLK_ASSIGN, 256, 0, stream>>>(zb, cmat, cn2, partial);
    reduce_kernel<<<NRED, 256, 0, stream>>>(partial, reduced);
    update_kernel<<<1, 256, 0, stream>>>(reduced, centers, cmat, cn2, shift);
  }
  prep_kernel<<<1, 256, 0, stream>>>(u, w, cmat, uhat, wuhat);
  final_kernel<<<NBLK_FINAL, 256, 0, stream>>>(z, zb, cmat, cn2, uhat, wuhat, b,
                                               (float*)d_out);
}
